// EventDenoisingMamba_59502476919111
// MI455X (gfx1250) — matrix operand order repaired and checked by reading, not compiled, not run on hardware
//
#include <hip/hip_runtime.h>

// ---------------- problem constants ----------------
#define BB 8
#define SS 8192
#define FF 11
#define DMODEL 128
#define DSTATE 16
#define DCONV 4
#define DINNER 256
#define DTRANK 8
#define NLAYERS 4
#define MM (BB * SS)              // 65536 tokens
#define XPN (DTRANK + 2 * DSTATE) // 40
#define XPNPAD 64                 // x_proj N padded to one 64-wide tile
#define NC 64                     // scan chunks
#define CL (SS / NC)              // 128 steps per chunk

typedef __attribute__((ext_vector_type(16))) __bf16 v16bf;
typedef __attribute__((ext_vector_type(8)))  __bf16 v8bf;
typedef __attribute__((ext_vector_type(8)))  float  v8f;

// ---------------- WMMA fragment helpers (CDNA5 16x16x32 bf16) ----------------
// A: 16x32 bf16 row-major (lda = K). ISA layout:
//   lanes 0-15: M=lane, K=0..7 & 16..23 ; lanes 16-31: M=lane-16, K=8..15 & 24..31
__device__ __forceinline__ v16bf load_a_frag(const __bf16* __restrict__ A, int lda,
                                             int m0, int k0, int lane) {
    int row = m0 + (lane & 15);
    int klo = (lane >> 4) * 8;
    const __bf16* p = A + (size_t)row * lda + k0 + klo;
    v8bf lo = *(const v8bf*)p;          // K = klo .. klo+7       (16B aligned)
    v8bf hi = *(const v8bf*)(p + 16);   // K = 16+klo .. 16+klo+7
    v16bf a;
#pragma unroll
    for (int i = 0; i < 8; ++i) { a[i] = lo[i]; a[8 + i] = hi[i]; }
    return a;
}

// B: 32x16 (KxN) fragment from pre-transposed bf16 weights WT (Npad x K row-major).
// ISA layout: lanes 0-15: col=lane, K=k0+0..7 & 16..23 ; lanes 16-31: col=lane-16, K=k0+8..15 & 24..31
// -> per lane two 8-bf16 runs at WT[col*K + kb] and +16, two aligned 16B loads, no guards.
__device__ __forceinline__ v16bf load_b_frag(const __bf16* __restrict__ WT, int K,
                                             int k0, int n0, int lane) {
    int col = n0 + (lane & 15);
    int kb  = k0 + (lane >> 4) * 8;
    const __bf16* p = WT + (size_t)col * K + kb;
    v8bf lo = *(const v8bf*)p;
    v8bf hi = *(const v8bf*)(p + 16);
    v16bf b;
#pragma unroll
    for (int i = 0; i < 8; ++i) { b[i] = lo[i]; b[8 + i] = hi[i]; }
    return b;
}

// ---------------- weight prep: W (K x N f32) -> WT (Npad x K bf16), zero padded ----------------
__global__ void prep_weights_kernel(const float* __restrict__ W, __bf16* __restrict__ WT,
                                    int K, int N, int Npad) {
    size_t idx = (size_t)blockIdx.x * blockDim.x + threadIdx.x; // over Npad*K
    if (idx >= (size_t)Npad * K) return;
    int col = (int)(idx / K);
    int kk  = (int)(idx % K);
    float v = (col < N) ? W[(size_t)kk * N + col] : 0.0f;
    WT[idx] = (__bf16)v;
}

// ---------------- generic WMMA GEMM: C[M,N] = Abf[M,K] @ WTbf[Npad,K]^T ----------------
// one wave -> 16(M) x 64(N) strip, 4 accumulators, K step 32.
__global__ void gemm_wmma_bf16(const __bf16* __restrict__ A,
                               const __bf16* __restrict__ WT,
                               float* __restrict__ Cf,      // may be null; stride N
                               __bf16* __restrict__ Cbf,    // may be null; stride N
                               int M, int N, int K, int n_tiles_n) {
    int wave = (int)((blockIdx.x * blockDim.x + threadIdx.x) >> 5);
    int lane = threadIdx.x & 31;
    int mt = wave / n_tiles_n;
    int nt = wave % n_tiles_n;
    int m0 = mt * 16;
    if (m0 >= M) return;                 // wave-uniform; EXEC stays all-ones
    int n0 = nt * 64;

    v8f acc0 = {}, acc1 = {}, acc2 = {}, acc3 = {};
    for (int k = 0; k < K; k += 32) {
        // speculative prefetch of next K slice (silently dropped if OOB)
        __builtin_prefetch(A + (size_t)(m0 + (lane & 15)) * K + k + 32, 0, 1);
        __builtin_prefetch(WT + (size_t)(n0 + (lane & 15)) * K + k + 32, 0, 1);

        v16bf a  = load_a_frag(A, K, m0, k, lane);
        v16bf b0 = load_b_frag(WT, K, k, n0 +  0, lane);
        v16bf b1 = load_b_frag(WT, K, k, n0 + 16, lane);
        v16bf b2 = load_b_frag(WT, K, k, n0 + 32, lane);
        v16bf b3 = load_b_frag(WT, K, k, n0 + 48, lane);
        acc0 = __builtin_amdgcn_wmma_f32_16x16x32_bf16(false, a, false, b0, (short)0, acc0, false, false);
        acc1 = __builtin_amdgcn_wmma_f32_16x16x32_bf16(false, a, false, b1, (short)0, acc1, false, false);
        acc2 = __builtin_amdgcn_wmma_f32_16x16x32_bf16(false, a, false, b2, (short)0, acc2, false, false);
        acc3 = __builtin_amdgcn_wmma_f32_16x16x32_bf16(false, a, false, b3, (short)0, acc3, false, false);
    }
    // C/D layout: lanes 0-15 col=lane rows m0+0..7 ; lanes 16-31 col=lane-16 rows m0+8..15
    int colb  = lane & 15;
    int rbase = m0 + (lane >> 4) * 8;
    v8f accs[4] = {acc0, acc1, acc2, acc3};
#pragma unroll
    for (int j = 0; j < 4; ++j) {
        int col = n0 + 16 * j + colb;
        if (col >= N) continue;          // only trims padded cols (x_proj)
#pragma unroll
        for (int i = 0; i < 8; ++i) {
            size_t idx = (size_t)(rbase + i) * N + col;
            float v = accs[j][i];
            if (Cf)  Cf[idx]  = v;
            if (Cbf) Cbf[idx] = (__bf16)v;
        }
    }
}

// ---------------- embedding: x = feat @ emb_w + emb_b (bf16 mirror only) ----------------
__global__ void embed_kernel(const float* __restrict__ feat, const float* __restrict__ ew,
                             const float* __restrict__ eb, __bf16* __restrict__ xbf) {
    size_t idx = (size_t)blockIdx.x * blockDim.x + threadIdx.x; // over M*128
    int d = (int)(idx & (DMODEL - 1));
    size_t m = idx >> 7;
    float acc = eb[d];
    const float* f = feat + m * FF;
#pragma unroll
    for (int j = 0; j < FF; ++j) acc += f[j] * ew[j * DMODEL + d];
    xbf[idx] = (__bf16)acc;
}

// ---------------- causal depthwise conv(4) + SiLU ----------------
__global__ void conv_silu_kernel(const float* __restrict__ xz,   // (M,512), u = cols 0..255
                                 const float* __restrict__ cw,   // (256,4)
                                 const float* __restrict__ cb,   // (256)
                                 float* __restrict__ u, __bf16* __restrict__ ubf) {
    size_t idx = (size_t)blockIdx.x * blockDim.x + threadIdx.x;  // over M*256
    int d = (int)(idx & (DINNER - 1));
    size_t m = idx >> 8;
    int s = (int)(m & (SS - 1));
    float acc = cb[d];
    const float* w = cw + d * DCONV;
#pragma unroll
    for (int j = 0; j < DCONV; ++j) {
        int ss = s - (DCONV - 1) + j;
        if (ss >= 0) acc += w[j] * xz[(m - (DCONV - 1) + j) * 512 + d];
    }
    float v = acc / (1.0f + __expf(-acc));   // SiLU
    u[idx] = v;
    ubf[idx] = (__bf16)v;
}

// ---------------- delta = softplus(dt @ dt_w + dt_b) ----------------
__global__ void delta_kernel(const float* __restrict__ xdbl,   // (M,40)
                             const float* __restrict__ dtw,    // (8,256)
                             const float* __restrict__ dtb,    // (256)
                             float* __restrict__ delta) {
    size_t idx = (size_t)blockIdx.x * blockDim.x + threadIdx.x;  // over M*256
    int d = (int)(idx & (DINNER - 1));
    size_t m = idx >> 8;
    const float* row = xdbl + m * XPN;
    float acc = dtb[d];
#pragma unroll
    for (int r = 0; r < DTRANK; ++r) acc += row[r] * dtw[r * DINNER + d];
    delta[idx] = (acc > 20.0f) ? acc : log1pf(__expf(acc));
}

// ---------------- scan phase A: per-chunk (prod dA, folded dB) ----------------
__global__ void scan_chunk_reduce(const float* __restrict__ delta, const float* __restrict__ u,
                                  const float* __restrict__ xdbl, const float* __restrict__ A_log,
                                  float* __restrict__ P, float* __restrict__ Q) {
    size_t tid = (size_t)blockIdx.x * blockDim.x + threadIdx.x;  // over NC*B*256
    int d = (int)(tid & (DINNER - 1));
    int bc = (int)(tid >> 8);
    int b = bc % BB, c = bc / BB;
    float a[DSTATE], Pr[DSTATE], Qr[DSTATE];
#pragma unroll
    for (int n = 0; n < DSTATE; ++n) {
        a[n] = -__expf(A_log[d * DSTATE + n]);
        Pr[n] = 1.0f; Qr[n] = 0.0f;
    }
    size_t mbase = (size_t)b * SS + (size_t)c * CL;
    for (int t = 0; t < CL; ++t) {
        size_t m = mbase + t;
        float dt = delta[m * DINNER + d];
        float du = dt * u[m * DINNER + d];
        const float* Brow = xdbl + m * XPN + DTRANK;
#pragma unroll
        for (int n = 0; n < DSTATE; ++n) {
            float dA = __expf(dt * a[n]);
            Pr[n] *= dA;
            Qr[n] = Qr[n] * dA + du * Brow[n];
        }
    }
    size_t o = (((size_t)c * BB + b) * DINNER + d) * DSTATE;
#pragma unroll
    for (int n = 0; n < DSTATE; ++n) { P[o + n] = Pr[n]; Q[o + n] = Qr[n]; }
}

// ---------------- scan phase B: prefix over chunk summaries ----------------
__global__ void scan_prefix(const float* __restrict__ P, const float* __restrict__ Q,
                            float* __restrict__ Hinit) {
    size_t tid = (size_t)blockIdx.x * blockDim.x + threadIdx.x;  // over B*256*16
    const size_t stride = (size_t)BB * DINNER * DSTATE;
    float h = 0.0f;
    for (int c = 0; c < NC; ++c) {
        size_t o = (size_t)c * stride + tid;
        Hinit[o] = h;
        h = P[o] * h + Q[o];
    }
}

// ---------------- scan phase C: replay chunk with restored state, emit y ----------------
__global__ void scan_chunk_apply(const float* __restrict__ delta, const float* __restrict__ u,
                                 const float* __restrict__ xdbl, const float* __restrict__ A_log,
                                 const float* __restrict__ Hinit,
                                 float* __restrict__ y512 /* y -> cols 0..255 of xz */) {
    size_t tid = (size_t)blockIdx.x * blockDim.x + threadIdx.x;  // over NC*B*256
    int d = (int)(tid & (DINNER - 1));
    int bc = (int)(tid >> 8);
    int b = bc % BB, c = bc / BB;
    float a[DSTATE], h[DSTATE];
    size_t o = (((size_t)c * BB + b) * DINNER + d) * DSTATE;
#pragma unroll
    for (int n = 0; n < DSTATE; ++n) {
        a[n] = -__expf(A_log[d * DSTATE + n]);
        h[n] = Hinit[o + n];
    }
    size_t mbase = (size_t)b * SS + (size_t)c * CL;
    for (int t = 0; t < CL; ++t) {
        size_t m = mbase + t;
        float dt = delta[m * DINNER + d];
        float du = dt * u[m * DINNER + d];
        const float* Brow = xdbl + m * XPN + DTRANK;
        const float* Crow = xdbl + m * XPN + DTRANK + DSTATE;
        float acc = 0.0f;
#pragma unroll
        for (int n = 0; n < DSTATE; ++n) {
            float dA = __expf(dt * a[n]);
            h[n] = h[n] * dA + du * Brow[n];
            acc += h[n] * Crow[n];
        }
        y512[m * 512 + d] = acc;
    }
}

// ---------------- y = (y + u*D) * silu(z), bf16 out for out_proj ----------------
__global__ void ypost_kernel(const float* __restrict__ xz,   // y in cols 0..255, z in 256..511
                             const float* __restrict__ u, const float* __restrict__ Dp,
                             __bf16* __restrict__ ybf) {
    size_t idx = (size_t)blockIdx.x * blockDim.x + threadIdx.x;  // over M*256
    int d = (int)(idx & (DINNER - 1));
    size_t m = idx >> 8;
    float yv = xz[m * 512 + d] + u[idx] * Dp[d];
    float z  = xz[m * 512 + 256 + d];
    float zs = z / (1.0f + __expf(-z));
    ybf[idx] = (__bf16)(yv * zs);
}

// ---------------- head: sigmoid(x @ head_w + head_b) ----------------
__global__ void head_kernel(const float* __restrict__ x, const float* __restrict__ hw,
                            const float* __restrict__ hb, float* __restrict__ out) {
    __shared__ float red[DMODEL];
    int m = blockIdx.x, t = threadIdx.x;
    red[t] = x[(size_t)m * DMODEL + t] * hw[t];
    __syncthreads();
    for (int off = DMODEL / 2; off > 0; off >>= 1) {
        if (t < off) red[t] += red[t + off];
        __syncthreads();
    }
    if (t == 0) {
        float v = red[0] + hb[0];
        out[m] = 1.0f / (1.0f + __expf(-v));
    }
}

// ---------------- host side ----------------
extern "C" void kernel_launch(void* const* d_in, const int* in_sizes, int n_in,
                              void* d_out, int out_size, void* d_ws, size_t ws_size,
                              hipStream_t stream) {
    const float* feat   = (const float*)d_in[0];
    const float* emb_w  = (const float*)d_in[1];
    const float* emb_b  = (const float*)d_in[2];
    const float* in_w   = (const float*)d_in[3];   // (4,128,512)
    const float* conv_w = (const float*)d_in[4];   // (4,256,4)
    const float* conv_b = (const float*)d_in[5];   // (4,256)
    const float* xp_w   = (const float*)d_in[6];   // (4,256,40)
    const float* dt_w   = (const float*)d_in[7];   // (4,8,256)
    const float* dt_b   = (const float*)d_in[8];   // (4,256)
    const float* A_log  = (const float*)d_in[9];   // (4,256,16)
    const float* Dp     = (const float*)d_in[10];  // (4,256)
    const float* out_w  = (const float*)d_in[11];  // (4,256,128)
    const float* head_w = (const float*)d_in[12];  // (128,1)
    const float* head_b = (const float*)d_in[13];  // (1)
    float* out = (float*)d_out;

    // workspace carve-out (256B aligned)
    char* p = (char*)d_ws;
    auto alloc = [&](size_t bytes) -> void* {
        void* r = (void*)p;
        p += (bytes + 255) & ~(size_t)255;
        return r;
    };
    __bf16* xbf   = (__bf16*)alloc((size_t)MM * DMODEL * 2);
    float*  xz    = (float*) alloc((size_t)MM * 512 * 4);
    float*  u     = (float*) alloc((size_t)MM * DINNER * 4);
    __bf16* ubf   = (__bf16*)alloc((size_t)MM * DINNER * 2);  // reused as y_bf16
    float*  xdbl  = (float*) alloc((size_t)MM * XPN * 4);
    float*  delta = (float*) alloc((size_t)MM * DINNER * 4);
    float*  xf32  = (float*) alloc((size_t)MM * DMODEL * 4);
    size_t chsum  = (size_t)NC * BB * DINNER * DSTATE;        // 2M elems
    float*  P     = (float*) alloc(chsum * 4);
    float*  Q     = (float*) alloc(chsum * 4);
    float*  Hinit = (float*) alloc(chsum * 4);
    __bf16* wt_in = (__bf16*)alloc((size_t)512 * DMODEL * 2); // in_proj^T  (512 x 128)
    __bf16* wt_xp = (__bf16*)alloc((size_t)XPNPAD * DINNER * 2); // x_proj^T (64 x 256, padded)
    __bf16* wt_op = (__bf16*)alloc((size_t)DMODEL * DINNER * 2); // out_proj^T (128 x 256)
    (void)ws_size; (void)n_in; (void)in_sizes; (void)out_size;

    const dim3 blk(256);

    embed_kernel<<<(MM * DMODEL) / 256, blk, 0, stream>>>(feat, emb_w, emb_b, xbf);

    for (int l = 0; l < NLAYERS; ++l) {
        // weight prep (tiny): transpose + f32->bf16 + zero-pad
        prep_weights_kernel<<<(512 * DMODEL + 255) / 256, blk, 0, stream>>>(
            in_w + (size_t)l * DMODEL * 512, wt_in, DMODEL, 512, 512);
        prep_weights_kernel<<<(XPNPAD * DINNER + 255) / 256, blk, 0, stream>>>(
            xp_w + (size_t)l * DINNER * XPN, wt_xp, DINNER, XPN, XPNPAD);
        prep_weights_kernel<<<(DMODEL * DINNER + 255) / 256, blk, 0, stream>>>(
            out_w + (size_t)l * DINNER * DMODEL, wt_op, DINNER, DMODEL, DMODEL);

        // in_proj: (M,128)bf16 @ (128,512) -> xz f32
        gemm_wmma_bf16<<<(MM / 16) * 8 / 8, blk, 0, stream>>>(
            xbf, wt_in, xz, nullptr, MM, 512, DMODEL, 8);

        conv_silu_kernel<<<(MM * DINNER) / 256, blk, 0, stream>>>(
            xz, conv_w + (size_t)l * DINNER * DCONV, conv_b + (size_t)l * DINNER, u, ubf);

        // x_proj: (M,256)bf16 @ (256,40) -> xdbl f32 (padded 64-wide tile, store-guarded)
        gemm_wmma_bf16<<<(MM / 16) * 1 / 8, blk, 0, stream>>>(
            ubf, wt_xp, xdbl, nullptr, MM, XPN, DINNER, 1);

        delta_kernel<<<(MM * DINNER) / 256, blk, 0, stream>>>(
            xdbl, dt_w + (size_t)l * DTRANK * DINNER, dt_b + (size_t)l * DINNER, delta);

        const float* Al = A_log + (size_t)l * DINNER * DSTATE;
        scan_chunk_reduce<<<(NC * BB * DINNER) / 256, blk, 0, stream>>>(delta, u, xdbl, Al, P, Q);
        scan_prefix<<<(BB * DINNER * DSTATE) / 256, blk, 0, stream>>>(P, Q, Hinit);
        scan_chunk_apply<<<(NC * BB * DINNER) / 256, blk, 0, stream>>>(delta, u, xdbl, Al, Hinit, xz);

        ypost_kernel<<<(MM * DINNER) / 256, blk, 0, stream>>>(xz, u, Dp + (size_t)l * DINNER, ubf);

        // out_proj: (M,256)bf16 @ (256,128) -> xf32 + xbf (next layer's A operand)
        gemm_wmma_bf16<<<(MM / 16) * 2 / 8, blk, 0, stream>>>(
            ubf, wt_op, xf32, xbf, MM, DMODEL, DINNER, 2);
    }

    head_kernel<<<MM, dim3(DMODEL), 0, stream>>>(xf32, head_w, head_b, out);
}
